// UpThreeOffsetsConvShareWeights_87660282511814
// MI455X (gfx1250) — matrix operand order repaired and checked by reading, not compiled, not run on hardware
//
#include <hip/hip_runtime.h>
#include <stdint.h>

// ---------------------------------------------------------------------------
// MI455X / gfx1250 fused implementation.
//
//  out = relu(BN2( Wcomb @ concat_b relu(BN1( conv5x5x5_b(up2(x)) + b_def )) + b_comb ))
//
// conv5x5x5_b = effective dense kernel of the static-offset deformable conv.
// All GEMM work on v_wmma_f32_16x16x32_bf16 (wave32).
// NT=4: one wave owns a 64-wide x strip -> 24 WMMA per 20 b128 loads/lane.
// ---------------------------------------------------------------------------

typedef __attribute__((ext_vector_type(16))) __bf16 v16bf;
typedef __attribute__((ext_vector_type(8)))  float  v8f;

#define ZP     68        // padded spatial extent (64 + 2*2)
#define CIN    64
#define NCHUNK 250       // 125 taps * 2 channel-halves (K chunks of 32)
#define NT     4         // N-tiles of 16 x-positions per wave

__constant__ float c_OFF[81] = {
  1,1,1, 1,1,0, 1,1,-1, 1,0,1, 1,0,0, 1,0,-1, 1,-1,1, 1,-1,0, 1,-1,-1,
  0,1,1, 0,1,0, 0,1,-1, 0,0,1, 0,0,0, 0,0,-1, 0,-1,1, 0,-1,0, 0,-1,-1,
  -1,1,1, -1,1,0, -1,1,-1, -1,0,1, -1,0,0, -1,0,-1, -1,-1,1, -1,-1,0, -1,-1,-1
};
__constant__ float c_ALPHA[3] = {0.0f, 0.4f, 0.7f};

static __device__ __forceinline__ uint16_t f2bf(float f) {
  uint32_t u = __builtin_bit_cast(uint32_t, f);
  u += 0x7FFFu + ((u >> 16) & 1u);      // round-to-nearest-even
  return (uint16_t)(u >> 16);
}

static __device__ __forceinline__ v16bf ldFrag(const void* p) {
  union { float4 f[2]; v16bf v; } u;
  const float4* q = (const float4*)p;
  u.f[0] = q[0];
  u.f[1] = q[1];
  return u.v;
}

// --------------------------------------------------------------------------
// Prep 1: up2(x) + pad 2, fp32 -> bf16, channel-LAST layout xp[z][y][x][c].
// --------------------------------------------------------------------------
__global__ void upsample_pad_k(const float* __restrict__ x, uint16_t* __restrict__ xp) {
  size_t t = (size_t)blockIdx.x * blockDim.x + threadIdx.x;
  if (t >= (size_t)ZP * ZP * ZP * CIN) return;
  int c = (int)(t & 63);
  size_t vox = t >> 6;
  int xi = (int)(vox % ZP); vox /= ZP;
  int yi = (int)(vox % ZP);
  int zi = (int)(vox / ZP);
  float v = 0.0f;
  int z = zi - 2, y = yi - 2, xx = xi - 2;
  if ((unsigned)z < 64u && (unsigned)y < 64u && (unsigned)xx < 64u) {
    v = x[(((size_t)c * 32 + (z >> 1)) * 32 + (y >> 1)) * 32 + (xx >> 1)];
  }
  xp[t] = f2bf(v);
}

// --------------------------------------------------------------------------
// Prep 2: effective 5x5x5 kernels, BN1 scale folded, pre-swizzled into the
// 16-bit WMMA A layout (16x32 tile per K-chunk).  One thread per packed dword.
// afrag[((chunk*6 + s)*32 + lane)*8 + v],  s = branch*2 + mtile.
// A layout used here (contiguous K): lane<16 -> M=lane, K in {0..15};
// lane>=16 -> M=lane-16, K in {16..31}; 2 K per dword.
// --------------------------------------------------------------------------
__global__ void build_afrag_k(const float* __restrict__ w_def,
                              const float* __restrict__ bn1_g,
                              const float* __restrict__ bn1_v,
                              uint32_t* __restrict__ afrag) {
  int t = blockIdx.x * blockDim.x + threadIdx.x;
  if (t >= 6 * NCHUNK * 32 * 8) return;
  int v    = t & 7;
  int lane = (t >> 3) & 31;
  int cs   = t >> 8;
  int s    = cs % 6;
  int chunk = cs / 6;
  int M  = lane & 15;
  int hi = lane >> 4;
  int k0 = hi * 16 + v * 2;  // even K of the pair
  int b  = s >> 1, mt = s & 1;
  int o  = mt * 16 + M;
  int tap = chunk >> 1, chalf = chunk & 1;
  int c0 = chalf * 32 + k0;
  int pz = tap / 25, py = (tap / 5) % 5, px = tap % 5;
  float alpha = c_ALPHA[b];

  float a0 = 0.0f, a1 = 0.0f;
  for (int t27 = 0; t27 < 27; ++t27) {
    int kz = t27 / 9, ky = (t27 / 3) % 3, kx = t27 % 3;
    float sz = kz - 1 + alpha * c_OFF[t27 * 3 + 0]; float fz = floorf(sz); float wz = sz - fz;
    float sy = ky - 1 + alpha * c_OFF[t27 * 3 + 1]; float fy = floorf(sy); float wy = sy - fy;
    float sx = kx - 1 + alpha * c_OFF[t27 * 3 + 2]; float fx = floorf(sx); float wx = sx - fx;
    for (int cz = 0; cz < 2; ++cz) {
      float wzc = cz ? wz : 1.0f - wz;
      if (wzc == 0.0f || (int)fz + cz + 2 != pz) continue;
      for (int cy = 0; cy < 2; ++cy) {
        float wyc = cy ? wy : 1.0f - wy;
        if (wyc == 0.0f || (int)fy + cy + 2 != py) continue;
        for (int cx = 0; cx < 2; ++cx) {
          float wxc = cx ? wx : 1.0f - wx;
          if (wxc == 0.0f || (int)fx + cx + 2 != px) continue;
          float wt = wzc * wyc * wxc;
          a0 += wt * w_def[((size_t)o * 64 + c0)     * 27 + t27];
          a1 += wt * w_def[((size_t)o * 64 + c0 + 1) * 27 + t27];
        }
      }
    }
  }
  float s1 = bn1_g[o] * rsqrtf(bn1_v[o] + 1e-5f);
  afrag[t] = (uint32_t)f2bf(a0 * s1) | ((uint32_t)f2bf(a1 * s1) << 16);
}

// --------------------------------------------------------------------------
// Prep 3: A-fragments for the 1x1 combine (BN2 scale folded) + fused biases.
// --------------------------------------------------------------------------
__global__ void build_a2_bias_k(const float* __restrict__ w_comb,
                                const float* __restrict__ b_def,
                                const float* __restrict__ bn1_g, const float* __restrict__ bn1_b,
                                const float* __restrict__ bn1_m, const float* __restrict__ bn1_v,
                                const float* __restrict__ b_comb,
                                const float* __restrict__ bn2_g, const float* __restrict__ bn2_b,
                                const float* __restrict__ bn2_m, const float* __restrict__ bn2_v,
                                uint32_t* __restrict__ a2frag,
                                float* __restrict__ bias1, float* __restrict__ bias2) {
  int t = blockIdx.x * blockDim.x + threadIdx.x;
  if (t < 6 * 32 * 8) {
    int v = t & 7, lane = (t >> 3) & 31, s = t >> 8;
    int M = lane & 15, hi = lane >> 4;
    int k0 = hi * 16 + v * 2;
    int b = s >> 1, mt = s & 1;
    int o2 = mt * 16 + M;
    float s2 = bn2_g[o2] * rsqrtf(bn2_v[o2] + 1e-5f);
    float v0 = s2 * w_comb[o2 * 96 + b * 32 + k0];
    float v1 = s2 * w_comb[o2 * 96 + b * 32 + k0 + 1];
    a2frag[t] = (uint32_t)f2bf(v0) | ((uint32_t)f2bf(v1) << 16);
  } else if (t < 1536 + 32) {
    int o = t - 1536;
    float s1 = bn1_g[o] * rsqrtf(bn1_v[o] + 1e-5f);
    bias1[o] = b_def[o] * s1 + bn1_b[o] - bn1_m[o] * s1;
  } else if (t < 1536 + 64) {
    int o = t - 1568;
    float s2 = bn2_g[o] * rsqrtf(bn2_v[o] + 1e-5f);
    bias2[o] = b_comb[o] * s2 + bn2_b[o] - bn2_m[o] * s2;
  }
}

// --------------------------------------------------------------------------
// Main fused kernel: one wave32 per workgroup, one (z,y) row, 64-wide strip.
// 24 persistent accumulators: {3 branches x 2 m-tiles} x {4 N-tiles of 16 x}.
// K loop: 125 taps x 2 channel-halves -> 48 v_wmma per tap.
// Epilogue: bias1+ReLU -> bf16 LDS (B2 layout) -> 24 v_wmma for Wcomb/BN2.
// --------------------------------------------------------------------------
__global__ __launch_bounds__(32) void fused_conv_k(const uint16_t* __restrict__ xp,
                                                   const uint32_t* __restrict__ afrag,
                                                   const uint32_t* __restrict__ a2frag,
                                                   const float* __restrict__ bias1,
                                                   const float* __restrict__ bias2,
                                                   float* __restrict__ out) {
  const int lane = threadIdx.x;     // 0..31
  const int n    = lane & 15;       // B column (x position within tile)
  const int kh   = lane >> 4;       // which 16-wide K half this lane holds
  const int y    = blockIdx.x;
  const int z    = blockIdx.y;

  const v8f vzero = {0.f, 0.f, 0.f, 0.f, 0.f, 0.f, 0.f, 0.f};
  v8f acc[6][NT];
#pragma unroll
  for (int s = 0; s < 6; ++s)
#pragma unroll
    for (int nt = 0; nt < NT; ++nt) acc[s][nt] = vzero;

  for (int tap = 0; tap < 125; ++tap) {
    const int dz = tap / 25, dy = (tap / 5) % 5, dx = tap % 5;
    const uint16_t* bpBase =
        xp + (((size_t)(z + dz) * ZP + (y + dy)) * ZP + (dx + n)) * CIN + kh * 16;

    if (tap + 1 < 125) {  // prefetch next tap's input line into near cache
      const int t2 = tap + 1;
      const int dz2 = t2 / 25, dy2 = (t2 / 5) % 5, dx2 = t2 % 5;
      __builtin_prefetch(
          xp + (((size_t)(z + dz2) * ZP + (y + dy2)) * ZP + (dx2 + n)) * CIN, 0, 3);
      __builtin_prefetch(
          xp + (((size_t)(z + dz2) * ZP + (y + dy2)) * ZP + (dx2 + n + 32)) * CIN, 0, 3);
    }

#pragma unroll
    for (int chalf = 0; chalf < 2; ++chalf) {
      const int chunk = tap * 2 + chalf;
      const uint16_t* bp = bpBase + chalf * 32;
      v16bf bfr[NT];
#pragma unroll
      for (int nt = 0; nt < NT; ++nt) bfr[nt] = ldFrag(bp + nt * 16 * CIN);
      const uint32_t* ap = afrag + ((size_t)chunk * 6 * 32 + lane) * 8;
#pragma unroll
      for (int s = 0; s < 6; ++s) {
        v16bf a = ldFrag(ap + (size_t)s * 32 * 8);
#pragma unroll
        for (int nt = 0; nt < NT; ++nt) {
          acc[s][nt] = __builtin_amdgcn_wmma_f32_16x16x32_bf16(
              false, a, false, bfr[nt], (short)0, acc[s][nt], false, false);
        }
      }
    }
  }

  // ---- epilogue stage 1: bias + ReLU, stash h as bf16 in B2 layout --------
  // hL[nt][b][n][o] : K (=branch-local out channel o) contiguous per column.
  __shared__ __align__(16) uint16_t hL[NT * 3 * 16 * 32];
#pragma unroll
  for (int s = 0; s < 6; ++s) {
    const int b = s >> 1, mt = s & 1;
#pragma unroll
    for (int nt = 0; nt < NT; ++nt) {
#pragma unroll
      for (int r = 0; r < 8; ++r) {
        const int o = mt * 16 + kh * 8 + r;      // C/D layout: lanes>=16 hold M+8
        float h = acc[s][nt][r] + bias1[o];
        h = fmaxf(h, 0.0f);
        hL[(((nt * 3 + b) * 16 + n) * 32) + o] = f2bf(h);
      }
    }
  }
  __syncthreads();

  // ---- epilogue stage 2: 32x96 combine GEMM (K = 3 chunks of 32) ----------
#pragma unroll
  for (int nt = 0; nt < NT; ++nt) {
#pragma unroll
    for (int mt = 0; mt < 2; ++mt) {
      v8f acc2 = vzero;
#pragma unroll
      for (int b = 0; b < 3; ++b) {
        v16bf a2 = ldFrag(a2frag + (((b * 2 + mt) * 32 + lane) * 8));
        v16bf b2 = ldFrag(&hL[((nt * 3 + b) * 16 + n) * 32 + kh * 16]);
        acc2 = __builtin_amdgcn_wmma_f32_16x16x32_bf16(
            false, a2, false, b2, (short)0, acc2, false, false);
      }
#pragma unroll
      for (int r = 0; r < 8; ++r) {
        const int o2 = mt * 16 + kh * 8 + r;
        float yv = fmaxf(acc2[r] + bias2[o2], 0.0f);
        out[(((size_t)o2 * 64 + z) * 64 + y) * 64 + (nt * 16 + n)] = yv;
      }
    }
  }
}

// --------------------------------------------------------------------------
extern "C" void kernel_launch(void* const* d_in, const int* in_sizes, int n_in,
                              void* d_out, int out_size, void* d_ws, size_t ws_size,
                              hipStream_t stream) {
  (void)in_sizes; (void)n_in; (void)out_size; (void)ws_size;
  const float* x     = (const float*)d_in[0];
  const float* w_def = (const float*)d_in[1];
  const float* b_def = (const float*)d_in[2];
  const float* bn1_g = (const float*)d_in[3];
  const float* bn1_b = (const float*)d_in[4];
  const float* bn1_m = (const float*)d_in[5];
  const float* bn1_v = (const float*)d_in[6];
  const float* w_cmb = (const float*)d_in[7];
  const float* b_cmb = (const float*)d_in[8];
  const float* bn2_g = (const float*)d_in[9];
  const float* bn2_b = (const float*)d_in[10];
  const float* bn2_m = (const float*)d_in[11];
  const float* bn2_v = (const float*)d_in[12];

  uint8_t* ws = (uint8_t*)d_ws;
  const size_t XP_BYTES = (size_t)ZP * ZP * ZP * CIN * 2;          // 40,247,296
  const size_t AF_BYTES = (size_t)6 * NCHUNK * 32 * 8 * 4;         //  1,536,000
  const size_t A2_BYTES = (size_t)6 * 32 * 8 * 4;                  //      6,144
  uint16_t* xp    = (uint16_t*)ws;
  uint32_t* afrag = (uint32_t*)(ws + XP_BYTES);
  uint32_t* a2f   = (uint32_t*)(ws + XP_BYTES + AF_BYTES);
  float*    bias1 = (float*)(ws + XP_BYTES + AF_BYTES + A2_BYTES);
  float*    bias2 = bias1 + 32;

  upsample_pad_k<<<78608, 256, 0, stream>>>(x, xp);
  build_afrag_k<<<1500, 256, 0, stream>>>(w_def, bn1_g, bn1_v, afrag);
  build_a2_bias_k<<<7, 256, 0, stream>>>(w_cmb, b_def, bn1_g, bn1_b, bn1_m, bn1_v,
                                         b_cmb, bn2_g, bn2_b, bn2_m, bn2_v,
                                         a2f, bias1, bias2);

  dim3 grid(64, 64);             // 4096 workgroups, 1 wave32 each, 64-wide strip
  fused_conv_k<<<grid, 32, 0, stream>>>(xp, afrag, a2f, bias1, bias2, (float*)d_out);
}
